// QLSTMEnhancedQuantum_65481071406582
// MI455X (gfx1250) — matrix operand order repaired and checked by reading, not compiled, not run on hardware
//
#include <hip/hip_runtime.h>
#include <hip/hip_bf16.h>
#include <math.h>

// Problem dims (from reference): B=512, T=1024, D=64, N=8, G=4, COMB=72
#define BQ   512
#define TQ   1024
#define DQ   64
#define NQ   8
#define GQ   4
#define COMBQ 72
#define NOUT 32                    // G*N fused gate outputs
#define BT   (BQ * TQ)             // 524288 rows for the hoisted x-GEMM
#define RT   4                     // row tiles per wave (B-fragment reuse)

typedef __attribute__((ext_vector_type(16))) _Float16 v16h;
typedef __attribute__((ext_vector_type(8)))  float    v8f;

// Convert a float4 into 4 consecutive f16 elements of a fragment.
__device__ __forceinline__ void cvt4(v16h& frag, int i, const float4 v) {
    frag[i + 0] = (_Float16)v.x;
    frag[i + 1] = (_Float16)v.y;
    frag[i + 2] = (_Float16)v.z;
    frag[i + 3] = (_Float16)v.w;
}

// ---------------------------------------------------------------------------
// Stage 1: hoisted x-part GEMM: pre[m][c] = sum_{d<64} x[m][d]*W[c][d] + b[c]
// One wave handles RT row tiles x BOTH 16-col tiles:
//   - A fragment built once per row tile, feeds 2 WMMAs (both col tiles)
//   - B fragments (2 col tiles x 2 K-blocks) built once per wave
// All fragment loads are 16B global_load_b128 (lane K-runs are contiguous).
// ---------------------------------------------------------------------------
__global__ __launch_bounds__(256) void xgemm_kernel(
    const float* __restrict__ x,     // (B*T, 64) row-major
    const float* __restrict__ W,     // (32, 72): row c = W_gates[g][n][:], c=g*8+n
    const float* __restrict__ bias,  // (32)
    float* __restrict__ pre)         // (B*T, 32)
{
    const int wave  = (blockIdx.x * blockDim.x + threadIdx.x) >> 5;
    const int lane  = threadIdx.x & 31;
    const int mlane = lane & 15;
    const int hi    = lane >> 4;
    const int m0    = wave * (16 * RT);          // first row of this wave's tiles

    // ---- B fragments: contiguous-K order chosen here (A uses the same): lane owns column c = c0+mlane;
    // lanes 0-15 hold K 0..15, lanes 16-31 hold K 16..31 (VGPR v = K pair 2v).
    v16h bfrag[2][2];                            // [colTile][koBlock]
    float bval[2];
#pragma unroll
    for (int ct = 0; ct < 2; ++ct) {
        const int c = ct * 16 + mlane;
        bval[ct] = bias[c];
        const float* wrow = W + (size_t)c * COMBQ;
#pragma unroll
        for (int kb = 0; kb < 2; ++kb) {         // ko = 0, 32
            const float* p = wrow + kb * 32 + hi * 16;
#pragma unroll
            for (int j = 0; j < 4; ++j)          // 16 contiguous K values
                cvt4(bfrag[ct][kb], 4 * j, *(const float4*)(p + 4 * j));
        }
    }

#pragma unroll
    for (int rt = 0; rt < RT; ++rt) {
        const int mt = m0 + rt * 16;
        const int m  = mt + mlane;               // A-matrix row owned by this lane

        // ---- A fragment: 16x32 f16 layout matching B: per K-block the lane's
        // 16 values live at {ko+hi*16 .. +15} (contiguous K)
        // -> 4 float4 loads per block.
        const float* xrow = x + (size_t)m * DQ;
        v16h afrag[2];
#pragma unroll
        for (int kb = 0; kb < 2; ++kb) {
            const float* p = xrow + kb * 32 + hi * 16;
            cvt4(afrag[kb], 0,  *(const float4*)(p + 0));
            cvt4(afrag[kb], 4,  *(const float4*)(p + 4));
            cvt4(afrag[kb], 8,  *(const float4*)(p + 8));
            cvt4(afrag[kb], 12, *(const float4*)(p + 12));
        }

        // ---- both column tiles share the A fragment
#pragma unroll
        for (int ct = 0; ct < 2; ++ct) {
            v8f acc;
#pragma unroll
            for (int r = 0; r < 8; ++r) acc[r] = bval[ct];  // bias folded in
            acc = __builtin_amdgcn_wmma_f32_16x16x32_f16(
                false, afrag[0], false, bfrag[ct][0], (short)0, acc, false, false);
            acc = __builtin_amdgcn_wmma_f32_16x16x32_f16(
                false, afrag[1], false, bfrag[ct][1], (short)0, acc, false, false);

            // C layout: VGPR r, lanes0-15 -> M=mt+r, lanes16-31 -> M=mt+8+r;
            // lane's column is ct*16+mlane. Stores coalesce across lanes.
            float* outp = pre + (size_t)(mt + hi * 8) * NOUT + ct * 16 + mlane;
#pragma unroll
            for (int r = 0; r < 8; ++r)
                outp[(size_t)r * NOUT] = acc[r];
        }
    }
}

// ---------------------------------------------------------------------------
// Zero the regulariser accumulator (d_out is poisoned before timing).
// ---------------------------------------------------------------------------
__global__ void zero_scalar(float* p) { *p = 0.0f; }

// ---------------------------------------------------------------------------
// Stage 2: sequential recurrence. One batch row per lane; rows independent,
// so each workgroup marches t = 0..T-1 with no cross-WG sync. Activation
// buffer is L2-resident (67 MB < 192 MB L2).
// ---------------------------------------------------------------------------
__device__ __forceinline__ float fast_sigmoid(float v) {
    return 1.0f / (1.0f + __expf(-v));
}

__global__ __launch_bounds__(32) void recurrent_kernel(
    const float* __restrict__ pre,      // (B*T, 32): bias-included x-part lin
    const float* __restrict__ Wg,       // (32, 72): h-part cols 64..71
    const float* __restrict__ P,        // (4, 8, 8)
    const float* __restrict__ pb,       // (4, 8)
    const float* __restrict__ qparams,  // (4, 8, 4)
    const float* __restrict__ Wreg,     // (8, 8)
    const float* __restrict__ breg,     // (8)
    float* __restrict__ out,            // (B, T, 8)
    float* __restrict__ reg_out)        // scalar at d_out[B*T*8]
{
    __shared__ float sWh[NOUT][NQ];     // recurrent weights
    __shared__ float sP[GQ][NQ][NQ];
    __shared__ float sPb[NOUT];
    __shared__ float sCos[NOUT];        // cos_phi = cos(sum_j qparams)
    __shared__ float sWreg[NQ][NQ];
    __shared__ float sBreg[NQ];
    __shared__ float sRed[32];

    const int lane = threadIdx.x;       // blockDim.x == 32 (one wave)

    {
        const int cc = lane;            // 32 lanes cover 32 fused outputs
#pragma unroll
        for (int j = 0; j < NQ; ++j) sWh[cc][j] = Wg[cc * COMBQ + DQ + j];
        sPb[cc] = pb[cc];
        const float s = qparams[cc * 4 + 0] + qparams[cc * 4 + 1] +
                        qparams[cc * 4 + 2] + qparams[cc * 4 + 3];
        sCos[cc] = __cosf(s);
    }
    for (int idx = lane; idx < GQ * NQ * NQ; idx += 32) ((float*)sP)[idx] = P[idx];
    for (int idx = lane; idx < NQ * NQ; idx += 32)      ((float*)sWreg)[idx] = Wreg[idx];
    if (lane < NQ) sBreg[lane] = breg[lane];
    __syncthreads();

    const int b = blockIdx.x * 32 + lane;     // one batch row per lane
    float hx[NQ], cx[NQ];
#pragma unroll
    for (int k = 0; k < NQ; ++k) { hx[k] = 0.0f; cx[k] = 0.0f; }
    float regacc = 0.0f;

    const float* prow = pre + (size_t)b * TQ * NOUT;
    float*       orow = out + (size_t)b * TQ * NQ;

    for (int t = 0; t < TQ; ++t) {
        const float* pv = prow + t * NOUT;
        if (t + 1 < TQ)                       // gfx1250 global_prefetch_b8
            __builtin_prefetch(prow + (t + 1) * NOUT, 0, 3);

        float lin[NOUT];
#pragma unroll
        for (int c4 = 0; c4 < NOUT / 4; ++c4) {
            const float4 v = *(const float4*)(pv + c4 * 4);  // global_load_b128
            lin[c4 * 4 + 0] = v.x; lin[c4 * 4 + 1] = v.y;
            lin[c4 * 4 + 2] = v.z; lin[c4 * 4 + 3] = v.w;
        }
        // h-part of the fused gate linear: K = 8
#pragma unroll
        for (int c = 0; c < NOUT; ++c) {
            float a = lin[c];
#pragma unroll
            for (int j = 0; j < NQ; ++j) a = fmaf(hx[j], sWh[c][j], a);
            lin[c] = a;
        }
        // QLayer: qp = lin @ P^T + pb ; z = cumprod(cos(qp) * cos_phi)
        float z[GQ][NQ];
#pragma unroll
        for (int g = 0; g < GQ; ++g) {
            float zc = 1.0f;
#pragma unroll
            for (int m = 0; m < NQ; ++m) {
                float q = sPb[g * NQ + m];
#pragma unroll
                for (int n = 0; n < NQ; ++n) q = fmaf(lin[g * NQ + n], sP[g][m][n], q);
                zc *= __cosf(q) * sCos[g * NQ + m];
                z[g][m] = zc;
            }
        }
        // LSTM cell + residual (residual = forget-gate linear, g=0)
#pragma unroll
        for (int k = 0; k < NQ; ++k) {
            const float f  = fast_sigmoid(z[0][k]);
            const float i  = fast_sigmoid(z[1][k]);
            const float gg = tanhf(z[2][k]);
            const float o  = fast_sigmoid(z[3][k]);
            cx[k] = f * cx[k] + i * gg;
            hx[k] = o * tanhf(cx[k]) + lin[k];
            orow[t * NQ + k] = hx[k];
        }
        // QRegulariser: 1 - prod cos^2(0.5 * (hx @ Wreg^T + breg))
        float fid = 1.0f;
#pragma unroll
        for (int m = 0; m < NQ; ++m) {
            float hp = sBreg[m];
#pragma unroll
            for (int k = 0; k < NQ; ++k) hp = fmaf(hx[k], sWreg[m][k], hp);
            const float cc = __cosf(0.5f * hp);
            fid *= cc * cc;
        }
        regacc += 1.0f - fid;
    }

    // Reduce the per-row regulariser sums across the wave, one atomic per WG.
    sRed[lane] = regacc;
    __syncthreads();
    if (lane == 0) {
        float s = 0.0f;
#pragma unroll
        for (int l = 0; l < 32; ++l) s += sRed[l];
        atomicAdd(reg_out, s * (1.0f / ((float)BQ * (float)TQ)));
    }
}

// ---------------------------------------------------------------------------
// Launch
// ---------------------------------------------------------------------------
extern "C" void kernel_launch(void* const* d_in, const int* in_sizes, int n_in,
                              void* d_out, int out_size, void* d_ws, size_t ws_size,
                              hipStream_t stream) {
    const float* x       = (const float*)d_in[0];  // (B, T, 64)
    const float* W_gates = (const float*)d_in[1];  // (4, 8, 72) == (32, 72)
    const float* b_gates = (const float*)d_in[2];  // (4, 8)    == (32)
    const float* P_gates = (const float*)d_in[3];  // (4, 8, 8)
    const float* pb      = (const float*)d_in[4];  // (4, 8)
    const float* qparams = (const float*)d_in[5];  // (4, 8, 4)
    const float* W_reg   = (const float*)d_in[6];  // (8, 8)
    const float* b_reg   = (const float*)d_in[7];  // (8)

    float* out     = (float*)d_out;                // (B, T, 8) + 1 scalar
    float* reg_out = out + (size_t)BQ * TQ * NQ;
    float* pre     = (float*)d_ws;                 // (B*T, 32) = 64 MiB scratch

    // Stage 1: 8192 waves (64 rows x 32 cols each), 8 waves per 256-thd block.
    const int nWaves  = BT / (16 * RT);
    const int nBlocks = nWaves / 8;
    xgemm_kernel<<<nBlocks, 256, 0, stream>>>(x, W_gates, b_gates, pre);

    // Stage 2: init scalar, then 16 single-wave WGs (32 batch rows each).
    zero_scalar<<<1, 1, 0, stream>>>(reg_out);
    recurrent_kernel<<<BQ / 32, 32, 0, stream>>>(
        pre, W_gates, P_gates, pb, qparams, W_reg, b_reg, out, reg_out);
}
